// CausalAdapter_64244120814028
// MI455X (gfx1250) — compile-verified, matrix operand order checked by reading, not run on hardware
//
#include <hip/hip_runtime.h>
#include <hip/hip_bf16.h>
#include <math.h>

// Shapes
#define B_  64
#define D_  768
#define H_  8
#define DH_ 128
#define NQ_ 256
#define NK_ 256

typedef __attribute__((ext_vector_type(16))) _Float16 v16h;
typedef __attribute__((ext_vector_type(8)))  _Float16 h8;
typedef __attribute__((ext_vector_type(8)))  float    v8f;
typedef __attribute__((ext_vector_type(4)))  float    f32x4;

// ---------------------------------------------------------------------------
// block-wide sum reduction (blockDim.x == 256)
__device__ __forceinline__ float block_sum(float v, float* sred) {
    int t = threadIdx.x;
    sred[t] = v;
    __syncthreads();
    #pragma unroll
    for (int s = 128; s > 0; s >>= 1) {
        if (t < s) sred[t] += sred[t + s];
        __syncthreads();
    }
    float r = sred[0];
    __syncthreads();
    return r;
}

__device__ __forceinline__ float gelu_exact(float x) {
    return 0.5f * x * (1.0f + erff(x * 0.70710678118654752f));
}
__device__ __forceinline__ float silu(float x) {
    return x / (1.0f + __expf(-x));
}

// ---------------------------------------------------------------------------
// Kernel 1: pooled_effect[b,d] = mean_q tok_effect[b,q,d]   (grid=B, block=256)
// tok_effect is streamed exactly once -> non-temporal loads.
__global__ void pool_kernel(const float* __restrict__ tok_effect,
                            float* __restrict__ pooled) {
    int b = blockIdx.x, t = threadIdx.x;
    #pragma unroll
    for (int jj = 0; jj < 3; ++jj) {
        int j = t + jj * 256;
        float acc = 0.0f;
        const float* base = tok_effect + (size_t)b * NQ_ * D_ + j;
        #pragma unroll 4
        for (int q = 0; q < NQ_; ++q)
            acc += __builtin_nontemporal_load(base + (size_t)q * D_);
        pooled[(size_t)b * D_ + j] = acc * (1.0f / NQ_);
    }
}

// ---------------------------------------------------------------------------
// Kernel 2: full context pipeline, one block per batch row (grid=B, block=256)
__global__ void ctx_kernel(
    const float* __restrict__ emb_cause, const float* __restrict__ emb_action,
    const float* __restrict__ fuse_ln_g, const float* __restrict__ fuse_ln_b,
    const float* __restrict__ fuse_w1,   const float* __restrict__ fuse_b1,
    const float* __restrict__ fuse_w2,   const float* __restrict__ fuse_b2,
    const float* __restrict__ delta_ln_g,const float* __restrict__ delta_ln_b,
    const float* __restrict__ delta_w,   const float* __restrict__ delta_b,
    const float* __restrict__ qg_w,      const float* __restrict__ qg_b,
    const float* __restrict__ kg_w,      const float* __restrict__ kg_b,
    const float* __restrict__ eff_ln_g,  const float* __restrict__ eff_ln_b,
    const float* __restrict__ eff_w1,    const float* __restrict__ eff_b1,
    const float* __restrict__ eff_w2,    const float* __restrict__ eff_b2,
    const float* __restrict__ u_w,
    const float* __restrict__ pooled,
    float* __restrict__ out_qgate, float* __restrict__ out_kgate,
    float* __restrict__ out_eshift, float* __restrict__ out_delta,
    float* __restrict__ out_effp,  float* __restrict__ ws_u) {

    __shared__ float s_a[2 * D_];   // 1536: ln(ctx_in), later [lnd | lne]/eh
    __shared__ float s_b[D_];       // h, later e_shifted
    __shared__ float s_c[D_];       // ctx
    __shared__ float s_red[256];

    int b = blockIdx.x, t = threadIdx.x;

    // --- load ctx_in = [emb_cause | emb_action] -----------------------------
    for (int i = t; i < 2 * D_; i += 256)
        s_a[i] = (i < D_) ? emb_cause[(size_t)b * D_ + i]
                          : emb_action[(size_t)b * D_ + (i - D_)];
    __syncthreads();

    // --- LN over 1536 -------------------------------------------------------
    {
        float ls = 0.0f;
        for (int i = t; i < 2 * D_; i += 256) ls += s_a[i];
        float mean = block_sum(ls, s_red) * (1.0f / (2 * D_));
        float lv = 0.0f;
        for (int i = t; i < 2 * D_; i += 256) { float d = s_a[i] - mean; lv += d * d; }
        float var = block_sum(lv, s_red) * (1.0f / (2 * D_));
        float inv = rsqrtf(var + 1e-5f);
        for (int i = t; i < 2 * D_; i += 256)
            s_a[i] = (s_a[i] - mean) * inv * fuse_ln_g[i] + fuse_ln_b[i];
    }
    __syncthreads();

    // --- h = GELU(ln @ fuse_w1 + b1) ---------------------------------------
    {
        float acc[3];
        #pragma unroll
        for (int jj = 0; jj < 3; ++jj) acc[jj] = fuse_b1[t + jj * 256];
        #pragma unroll 8
        for (int i = 0; i < 2 * D_; ++i) {
            float x = s_a[i];
            const float* wrow = fuse_w1 + (size_t)i * D_;
            #pragma unroll
            for (int jj = 0; jj < 3; ++jj) acc[jj] += x * wrow[t + jj * 256];
        }
        #pragma unroll
        for (int jj = 0; jj < 3; ++jj) s_b[t + jj * 256] = gelu_exact(acc[jj]);
    }
    __syncthreads();

    // --- ctx = h @ fuse_w2 + b2 --------------------------------------------
    {
        float acc[3];
        #pragma unroll
        for (int jj = 0; jj < 3; ++jj) acc[jj] = fuse_b2[t + jj * 256];
        #pragma unroll 8
        for (int i = 0; i < D_; ++i) {
            float x = s_b[i];
            const float* wrow = fuse_w2 + (size_t)i * D_;
            #pragma unroll
            for (int jj = 0; jj < 3; ++jj) acc[jj] += x * wrow[t + jj * 256];
        }
        #pragma unroll
        for (int jj = 0; jj < 3; ++jj) s_c[t + jj * 256] = acc[jj];
    }
    __syncthreads();

    // --- shared mean/var of ctx (delta-LN and eff-LN share them) -----------
    float mean2, inv2;
    {
        float ls = 0.0f;
        for (int i = t; i < D_; i += 256) ls += s_c[i];
        mean2 = block_sum(ls, s_red) * (1.0f / D_);
        float lv = 0.0f;
        for (int i = t; i < D_; i += 256) { float d = s_c[i] - mean2; lv += d * d; }
        float var2 = block_sum(lv, s_red) * (1.0f / D_);
        inv2 = rsqrtf(var2 + 1e-5f);
    }
    // lnd -> s_a[0:768], lne -> s_a[768:1536]
    for (int i = t; i < D_; i += 256) {
        float z = (s_c[i] - mean2) * inv2;
        s_a[i]       = z * delta_ln_g[i] + delta_ln_b[i];
        s_a[D_ + i]  = z * eff_ln_g[i]   + eff_ln_b[i];
    }
    __syncthreads();

    // --- delta_e, e_shifted -------------------------------------------------
    {
        float acc[3];
        #pragma unroll
        for (int jj = 0; jj < 3; ++jj) acc[jj] = delta_b[t + jj * 256];
        #pragma unroll 8
        for (int i = 0; i < D_; ++i) {
            float x = s_a[i];
            const float* wrow = delta_w + (size_t)i * D_;
            #pragma unroll
            for (int jj = 0; jj < 3; ++jj) acc[jj] += x * wrow[t + jj * 256];
        }
        #pragma unroll
        for (int jj = 0; jj < 3; ++jj) {
            int j = t + jj * 256;
            float de = acc[jj];
            float es = pooled[(size_t)b * D_ + j] + de;
            out_delta [(size_t)b * D_ + j] = de;
            out_eshift[(size_t)b * D_ + j] = es;
            s_b[j] = es;                 // h is dead; reuse as e_shifted
        }
    }
    __syncthreads();

    // --- eh = GELU(lne @ eff_w1 + b1)  -> overwrite s_a[0:768] -------------
    {
        float acc[3];
        #pragma unroll
        for (int jj = 0; jj < 3; ++jj) acc[jj] = eff_b1[t + jj * 256];
        #pragma unroll 8
        for (int i = 0; i < D_; ++i) {
            float x = s_a[D_ + i];
            const float* wrow = eff_w1 + (size_t)i * D_;
            #pragma unroll
            for (int jj = 0; jj < 3; ++jj) acc[jj] += x * wrow[t + jj * 256];
        }
        __syncthreads();
        #pragma unroll
        for (int jj = 0; jj < 3; ++jj) s_a[t + jj * 256] = gelu_exact(acc[jj]);
    }
    __syncthreads();

    // --- effect_pred = eh @ eff_w2 + b2 ------------------------------------
    {
        float acc[3];
        #pragma unroll
        for (int jj = 0; jj < 3; ++jj) acc[jj] = eff_b2[t + jj * 256];
        #pragma unroll 8
        for (int i = 0; i < D_; ++i) {
            float x = s_a[i];
            const float* wrow = eff_w2 + (size_t)i * D_;
            #pragma unroll
            for (int jj = 0; jj < 3; ++jj) acc[jj] += x * wrow[t + jj * 256];
        }
        #pragma unroll
        for (int jj = 0; jj < 3; ++jj)
            out_effp[(size_t)b * D_ + (t + jj * 256)] = acc[jj];
    }

    // --- gates (from ctx in s_c) and u (from e_shifted in s_b) -------------
    {
        float qa[4], ka[4], ua[4];
        #pragma unroll
        for (int jj = 0; jj < 4; ++jj) {
            int j = t + jj * 256;
            qa[jj] = qg_b[j];
            ka[jj] = kg_b[j];
            ua[jj] = 0.0f;
        }
        #pragma unroll 4
        for (int i = 0; i < D_; ++i) {
            float xc = s_c[i], xe = s_b[i];
            const float* qrow = qg_w + (size_t)i * (H_ * DH_);
            const float* krow = kg_w + (size_t)i * (H_ * DH_);
            const float* urow = u_w  + (size_t)i * (H_ * DH_);
            #pragma unroll
            for (int jj = 0; jj < 4; ++jj) {
                int j = t + jj * 256;
                qa[jj] += xc * qrow[j];
                ka[jj] += xc * krow[j];
                ua[jj] += xe * urow[j];
            }
        }
        #pragma unroll
        for (int jj = 0; jj < 4; ++jj) {
            int j = t + jj * 256;
            out_qgate[(size_t)b * (H_ * DH_) + j] = 1.0f + tanhf(silu(qa[jj]));
            out_kgate[(size_t)b * (H_ * DH_) + j] = 1.0f + tanhf(silu(ka[jj]));
            ws_u     [(size_t)b * (H_ * DH_) + j] = ua[jj];
        }
    }
}

// ---------------------------------------------------------------------------
// Kernel 3: p[b,h,d] = sum_dh u[b,h,dh] * v_w[d, h*128+dh]
// grid = B*H, block = 256 (each thread: 3 d-values, per-thread contiguous reads)
__global__ void pfold_kernel(const float* __restrict__ u,
                             const float* __restrict__ v_w,
                             float* __restrict__ p) {
    int bh = blockIdx.x;
    int b = bh >> 3, h = bh & 7;
    int t = threadIdx.x;
    __shared__ float s_u[DH_];
    if (t < DH_) s_u[t] = u[(size_t)(b * H_ + h) * DH_ + t];
    __syncthreads();
    #pragma unroll
    for (int jj = 0; jj < 3; ++jj) {
        int d = t + jj * 256;
        const f32x4* wrow = (const f32x4*)(v_w + (size_t)d * (H_ * DH_) + h * DH_);
        float acc = 0.0f;
        #pragma unroll 8
        for (int c = 0; c < DH_ / 4; ++c) {
            f32x4 w = wrow[c];
            acc += w.x * s_u[4 * c + 0] + w.y * s_u[4 * c + 1] +
                   w.z * s_u[4 * c + 2] + w.w * s_u[4 * c + 3];
        }
        p[(size_t)(b * H_ + h) * D_ + d] = acc;
    }
}

// ---------------------------------------------------------------------------
// Kernel 4: bias[b,h,k] = tok_cause[b,k,:] . p[b,h,:]  via WMMA f16->f32.
// One block per batch: 16 waves, wave w computes the 16-key tile w.
// p[b] staged to LDS as f16 ONCE per block (heads 8..15 zero-padded), padded
// row stride (776 halves) to spread LDS bank accesses. Two independent WMMA
// accumulator chains (even/odd K-chunks) to pipeline the matrix unit.
// M = keys, N = heads. EXEC all ones per wave as WMMA requires.
#define PROW 776  // 768 + 8 halves padding
__global__ void bias_wmma_kernel(const float* __restrict__ tok_cause,
                                 const float* __restrict__ p,
                                 float* __restrict__ bias) {
    __shared__ _Float16 s_p[16 * PROW];   // ~24.3 KB

    int b = blockIdx.x;
    int t = threadIdx.x;

    // --- stage p[b] -> LDS f16, zero-pad heads 8..15 ------------------------
    #pragma unroll
    for (int h = 0; h < 16; ++h) {
        for (int d = t; d < D_; d += 512) {
            s_p[h * PROW + d] = (h < H_)
                ? (_Float16)p[((size_t)(b * H_ + h)) * D_ + d]
                : (_Float16)0.0f;
        }
    }
    __syncthreads();

    int wave = t >> 5;          // key tile 0..15
    int lane = t & 31;
    int m  = lane & 15;         // key row for A / head col for B
    int hi = lane >> 4;         // K half-group selector

    const float* arow = tok_cause + ((size_t)(b * NK_ + wave * 16 + m)) * D_;

    v8f acc0 = {}, acc1 = {};
    for (int kc = 0; kc < D_ / 32; kc += 2) {    // 12 iters, 2 wmma each
        #pragma unroll
        for (int s = 0; s < 2; ++s) {
            int k0 = (kc + s) * 32 + hi * 8;     // halves 0..7
            int k1 = k0 + 16;                    // halves 8..15
            // A fragment: tok_cause streamed once -> non-temporal f32x4 loads
            f32x4 x0 = __builtin_nontemporal_load((const f32x4*)(arow + k0));
            f32x4 x1 = __builtin_nontemporal_load((const f32x4*)(arow + k0 + 4));
            f32x4 x2 = __builtin_nontemporal_load((const f32x4*)(arow + k1));
            f32x4 x3 = __builtin_nontemporal_load((const f32x4*)(arow + k1 + 4));
            v16h a;
            a[0] = (_Float16)x0.x;  a[1] = (_Float16)x0.y;
            a[2] = (_Float16)x0.z;  a[3] = (_Float16)x0.w;
            a[4] = (_Float16)x1.x;  a[5] = (_Float16)x1.y;
            a[6] = (_Float16)x1.z;  a[7] = (_Float16)x1.w;
            a[8] = (_Float16)x2.x;  a[9] = (_Float16)x2.y;
            a[10] = (_Float16)x2.z; a[11] = (_Float16)x2.w;
            a[12] = (_Float16)x3.x; a[13] = (_Float16)x3.y;
            a[14] = (_Float16)x3.z; a[15] = (_Float16)x3.w;
            // B fragment: two ds_load_b128 from staged f16 p
            h8 b0 = *(const h8*)&s_p[m * PROW + k0];
            h8 b1 = *(const h8*)&s_p[m * PROW + k1];
            v16h bf = __builtin_shufflevector(b0, b1,
                0, 1, 2, 3, 4, 5, 6, 7, 8, 9, 10, 11, 12, 13, 14, 15);
            if (s == 0)
                acc0 = __builtin_amdgcn_wmma_f32_16x16x32_f16(
                    false, a, false, bf, (short)0, acc0, false, false);
            else
                acc1 = __builtin_amdgcn_wmma_f32_16x16x32_f16(
                    false, a, false, bf, (short)0, acc1, false, false);
        }
    }

    // D layout: VGPR r -> M = r + (lane<16 ? 0 : 8), N = lane%16
    int h = m;
    if (h < H_) {
        #pragma unroll
        for (int r = 0; r < 8; ++r) {
            int k = wave * 16 + r + hi * 8;
            bias[((size_t)(b * H_ + h)) * NK_ + k] = acc0[r] + acc1[r];
        }
    }
}

// ---------------------------------------------------------------------------
// Kernel 5: logit_bias[b,h,q,k] = exp(logit_scale)/sqrt(DH) * bias[b,h,k]
// 134 MB of pure streaming stores -> float4 non-temporal.
// grid = 32768, block = 256 (8.39M float4 stores)
__global__ void broadcast_kernel(const float* __restrict__ bias,
                                 const float* __restrict__ logit_scale,
                                 float* __restrict__ out) {
    size_t idx = (size_t)blockIdx.x * blockDim.x + threadIdx.x; // float4 index
    int k4 = (int)(idx & 63);          // 256 k / 4
    size_t bhq = idx >> 6;
    size_t bh = bhq >> 8;              // drop q (256)
    float scale = __expf(logit_scale[0]) * 0.08838834764831845f; // 1/sqrt(128)
    const f32x4* b4 = (const f32x4*)bias;
    f32x4 v = b4[bh * 64 + k4] * scale;
    __builtin_nontemporal_store(v, ((f32x4*)out) + idx);
}

// ---------------------------------------------------------------------------
extern "C" void kernel_launch(void* const* d_in, const int* in_sizes, int n_in,
                              void* d_out, int out_size, void* d_ws, size_t ws_size,
                              hipStream_t stream) {
    const float* emb_cause  = (const float*)d_in[0];
    const float* emb_action = (const float*)d_in[1];
    const float* tok_cause  = (const float*)d_in[2];
    const float* tok_effect = (const float*)d_in[3];
    const float* fuse_ln_g  = (const float*)d_in[4];
    const float* fuse_ln_b  = (const float*)d_in[5];
    const float* fuse_w1    = (const float*)d_in[6];
    const float* fuse_b1    = (const float*)d_in[7];
    const float* fuse_w2    = (const float*)d_in[8];
    const float* fuse_b2    = (const float*)d_in[9];
    const float* delta_ln_g = (const float*)d_in[10];
    const float* delta_ln_b = (const float*)d_in[11];
    const float* delta_w    = (const float*)d_in[12];
    const float* delta_b    = (const float*)d_in[13];
    const float* qg_w       = (const float*)d_in[14];
    const float* qg_b       = (const float*)d_in[15];
    const float* kg_w       = (const float*)d_in[16];
    const float* kg_b       = (const float*)d_in[17];
    const float* eff_ln_g   = (const float*)d_in[18];
    const float* eff_ln_b   = (const float*)d_in[19];
    const float* eff_w1     = (const float*)d_in[20];
    const float* eff_b1     = (const float*)d_in[21];
    const float* eff_w2     = (const float*)d_in[22];
    const float* eff_b2     = (const float*)d_in[23];
    const float* u_w        = (const float*)d_in[24];
    const float* v_w        = (const float*)d_in[25];
    const float* logit_scale= (const float*)d_in[26];

    float* out = (float*)d_out;
    // tuple return order: q_gate, k_gate, e_shifted, delta_e, logit_bias, effect_pred
    float* out_qgate  = out;                                   // 64*8*128
    float* out_kgate  = out + 65536;                           // 64*8*128
    float* out_eshift = out + 131072;                          // 64*768
    float* out_delta  = out + 180224;                          // 64*768
    float* out_logit  = out + 229376;                          // 64*8*256*256
    float* out_effp   = out + 33783808;                        // 64*768

    float* ws     = (float*)d_ws;
    float* pooled = ws;                            // 64*768   = 49152
    float* ws_u   = pooled + (size_t)B_ * D_;      // 64*1024  = 65536
    float* ws_p   = ws_u + (size_t)B_ * H_ * DH_;  // 64*8*768 = 393216
    float* ws_bias= ws_p + (size_t)B_ * H_ * D_;   // 64*8*256 = 131072

    pool_kernel<<<B_, 256, 0, stream>>>(tok_effect, pooled);

    ctx_kernel<<<B_, 256, 0, stream>>>(
        emb_cause, emb_action,
        fuse_ln_g, fuse_ln_b, fuse_w1, fuse_b1, fuse_w2, fuse_b2,
        delta_ln_g, delta_ln_b, delta_w, delta_b,
        qg_w, qg_b, kg_w, kg_b,
        eff_ln_g, eff_ln_b, eff_w1, eff_b1, eff_w2, eff_b2,
        u_w, pooled,
        out_qgate, out_kgate, out_eshift, out_delta, out_effp, ws_u);

    pfold_kernel<<<B_ * H_, 256, 0, stream>>>(ws_u, v_w, ws_p);

    bias_wmma_kernel<<<B_, 512, 0, stream>>>(tok_cause, ws_p, ws_bias);

    broadcast_kernel<<<(B_ * H_ * NQ_ * NK_ / 4) / 256, 256, 0, stream>>>(
        ws_bias, logit_scale, out_logit);
}
